// RNN_55070070670224
// MI455X (gfx1250) — matrix operand order repaired and checked by reading, not compiled, not run on hardware
//
#include <hip/hip_runtime.h>
#include <math.h>

// Problem constants (from reference): V,E,H,L,S,B
#define Vv 10000
#define Ee 1024
#define Hh 1024
#define Ll 2
#define Ss 128
#define Bb 64

typedef __attribute__((ext_vector_type(16))) __bf16 v16bf;
typedef __attribute__((ext_vector_type(8)))  __bf16 v8bf;
typedef __attribute__((ext_vector_type(8)))  float  v8f;

__device__ __forceinline__ unsigned short f32_to_bf16_rne(float f) {
    union { float f; unsigned int u; } x;
    x.f = f;
    unsigned int u = x.u;
    u += 0x7FFFu + ((u >> 16) & 1u);   // round-to-nearest-even
    return (unsigned short)(u >> 16);
}

// ---------------------------------------------------------------------------
// Elementwise f32 -> bf16 (weight conversion, done once per launch)
// ---------------------------------------------------------------------------
__global__ void cvt_f32_bf16(const float* __restrict__ src,
                             unsigned short* __restrict__ dst, long n) {
    long i = (long)blockIdx.x * blockDim.x + threadIdx.x;
    if (i < n) dst[i] = f32_to_bf16_rne(src[i]);
}

// ---------------------------------------------------------------------------
// Embedding gather + scale(sqrt(E)=32) -> bf16 X0 [S*B, E]
// ---------------------------------------------------------------------------
__global__ void embed_scale(const int* __restrict__ tok,
                            const float* __restrict__ emb,
                            unsigned short* __restrict__ x0, long n) {
    long i = (long)blockIdx.x * blockDim.x + threadIdx.x;
    if (i >= n) return;
    long sb = i >> 10;              // / E (E == 1024)
    int  e  = (int)(i & 1023);
    int  t  = tok[sb];
    float v = emb[(long)t * Ee + e] * 32.0f;
    x0[i] = f32_to_bf16_rne(v);
}

// ---------------------------------------------------------------------------
// Initialize hidden state (parity-0 slot) from input hidden [L*B*H]
// ---------------------------------------------------------------------------
__global__ void init_hidden(const float* __restrict__ h0,
                            unsigned short* __restrict__ hbf,
                            float* __restrict__ hf32, int n) {
    int i = blockIdx.x * blockDim.x + threadIdx.x;
    if (i < n) {
        float v = h0[i];
        hbf[i]  = f32_to_bf16_rne(v);
        hf32[i] = v;
    }
}

// ---------------------------------------------------------------------------
// One RNN layer-step:  hnew = tanh(xin*Wih^T + hprev*Whh^T + bih + bhh)
// xin [B,1024] bf16, hprev [B,H] bf16, Wih [H,1024] bf16 rm, Whh [H,H] bf16 rm
// Grid: H/16 blocks of 128 threads (4 waves); wave w -> batch tile m=w,
// block -> one N (hidden-unit) tile. K = 1024 + 1024 via WMMA bf16 K=32.
// Latency-bound sequential part; weights (8MB bf16) stay resident in L2.
// ---------------------------------------------------------------------------
__global__ __launch_bounds__(128)
void rnn_step(const unsigned short* __restrict__ xin,
              const unsigned short* __restrict__ hprev,
              const unsigned short* __restrict__ wih,
              const unsigned short* __restrict__ whh,
              const float* __restrict__ bih,
              const float* __restrict__ bhh,
              unsigned short* __restrict__ hnew_bf,
              float* __restrict__ hnew_f32,
              unsigned short* __restrict__ h2all)   // nullptr for layer 0
{
    const int lane  = threadIdx.x & 31;
    const int wave  = threadIdx.x >> 5;
    const int nbase = blockIdx.x * 16;
    const int mbase = wave * 16;
    const bool lo   = lane < 16;
    const int l15   = lane & 15;
    // A-fragment per-lane K offsets (ISA 16-bit A 16x32 layout):
    //   lanes 0-15 : K in {0..7, 16..23}; lanes 16-31 : K in {8..15, 24..31}
    const int aoff0 = lo ? 0 : 8;
    const int aoff1 = lo ? 16 : 24;
    // B-fragment: lanes 0-15 hold K {0..7,16..23} of row N=lane, lanes 16-31 K {8..15,24..31}
    const int boff  = lo ? 0 : 8;

    v8f c = {};

    // ---- x * Wih^T (K = 1024) ----
    {
        const unsigned short* arow = xin + (size_t)(mbase + l15) * 1024;
        const unsigned short* brow = wih + (size_t)(nbase + l15) * 1024 + boff;
#pragma unroll 4
        for (int kt = 0; kt < 32; ++kt) {
            union { v16bf v; v8bf h[2]; } a;
            a.h[0] = *(const v8bf*)(arow + kt * 32 + aoff0);
            a.h[1] = *(const v8bf*)(arow + kt * 32 + aoff1);
            union { v16bf v; v8bf h[2]; } bu;
            bu.h[0] = *(const v8bf*)(brow + kt * 32);
            bu.h[1] = *(const v8bf*)(brow + kt * 32 + 16);
            v16bf b = bu.v;
            c = __builtin_amdgcn_wmma_f32_16x16x32_bf16(
                    false, a.v, false, b, (short)0, c, false, false);
        }
    }
    // ---- h * Whh^T (K = 1024) ----
    {
        const unsigned short* arow = hprev + (size_t)(mbase + l15) * Hh;
        const unsigned short* brow = whh   + (size_t)(nbase + l15) * Hh + boff;
#pragma unroll 4
        for (int kt = 0; kt < 32; ++kt) {
            union { v16bf v; v8bf h[2]; } a;
            a.h[0] = *(const v8bf*)(arow + kt * 32 + aoff0);
            a.h[1] = *(const v8bf*)(arow + kt * 32 + aoff1);
            union { v16bf v; v8bf h[2]; } bu;
            bu.h[0] = *(const v8bf*)(brow + kt * 32);
            bu.h[1] = *(const v8bf*)(brow + kt * 32 + 16);
            v16bf b = bu.v;
            c = __builtin_amdgcn_wmma_f32_16x16x32_bf16(
                    false, a.v, false, b, (short)0, c, false, false);
        }
    }

    const int n      = nbase + l15;
    const float bias = bih[n] + bhh[n];
    const int mofs   = lo ? 0 : 8;  // C/D: VGPR i -> M=i (lo lanes) / M=8+i (hi)
#pragma unroll
    for (int i = 0; i < 8; ++i) {
        int m   = mbase + mofs + i;
        float v = tanhf(c[i] + bias);
        size_t idx = (size_t)m * Hh + n;
        hnew_f32[idx] = v;
        unsigned short bv = f32_to_bf16_rne(v);
        hnew_bf[idx] = bv;
        if (h2all) h2all[idx] = bv;
    }
}

// ---------------------------------------------------------------------------
// Output projection: logits[8192, 10000] = H2all[8192,1024] * Wout^T + b_out
// Register-blocked: each wave computes a 2(M) x 5(N) block of 16x16 tiles
// (32 x 80 output). Per K-step: 2 A frags + 5 B frags feed 10 WMMAs
// (1.4 b128 loads per WMMA vs 4.0 unblocked -> ~3x less L2 traffic).
// M strips: 512/2=256, N strips: 625/5=125 -> 32000 wave-jobs, 8 waves/block.
// ---------------------------------------------------------------------------
#define MT 2
#define NT 5
__global__ __launch_bounds__(256)
void logits_gemm(const unsigned short* __restrict__ h2all,
                 const unsigned short* __restrict__ wout,
                 const float* __restrict__ bout,
                 float* __restrict__ out)
{
    const int lane = threadIdx.x & 31;
    const int wave = threadIdx.x >> 5;
    const int NSTRIPS = (Vv / 16) / NT;           // 125
    int job = blockIdx.x * 8 + wave;
    int ms  = job / NSTRIPS;                      // [0, 256)
    int ns  = job - ms * NSTRIPS;                 // [0, 125)
    const int mbase = ms * (MT * 16);
    const int nbase = ns * (NT * 16);

    const bool lo   = lane < 16;
    const int l15   = lane & 15;
    const int aoff0 = lo ? 0 : 8;
    const int aoff1 = lo ? 16 : 24;
    const int boff  = lo ? 0 : 8;

    const unsigned short* arow[MT];
#pragma unroll
    for (int mi = 0; mi < MT; ++mi)
        arow[mi] = h2all + (size_t)(mbase + mi * 16 + l15) * Hh;
    const unsigned short* brow[NT];
#pragma unroll
    for (int ni = 0; ni < NT; ++ni)
        brow[ni] = wout + (size_t)(nbase + ni * 16 + l15) * Hh + boff;

    v8f acc[MT][NT];
#pragma unroll
    for (int mi = 0; mi < MT; ++mi)
#pragma unroll
        for (int ni = 0; ni < NT; ++ni)
            acc[mi][ni] = (v8f){};

    for (int kt = 0; kt < 32; ++kt) {
        v16bf b[NT];
#pragma unroll
        for (int ni = 0; ni < NT; ++ni) {
            if (kt < 30) __builtin_prefetch(brow[ni] + (kt + 2) * 32, 0, 1);
            union { v16bf v; v8bf h[2]; } bu;
            bu.h[0] = *(const v8bf*)(brow[ni] + kt * 32);
            bu.h[1] = *(const v8bf*)(brow[ni] + kt * 32 + 16);
            b[ni] = bu.v;
        }
#pragma unroll
        for (int mi = 0; mi < MT; ++mi) {
            union { v16bf v; v8bf h[2]; } a;
            a.h[0] = *(const v8bf*)(arow[mi] + kt * 32 + aoff0);
            a.h[1] = *(const v8bf*)(arow[mi] + kt * 32 + aoff1);
#pragma unroll
            for (int ni = 0; ni < NT; ++ni) {
                acc[mi][ni] = __builtin_amdgcn_wmma_f32_16x16x32_bf16(
                        false, a.v, false, b[ni], (short)0, acc[mi][ni],
                        false, false);
            }
        }
    }

    const int mofs = lo ? 0 : 8;
#pragma unroll
    for (int mi = 0; mi < MT; ++mi) {
#pragma unroll
        for (int ni = 0; ni < NT; ++ni) {
            const int n     = nbase + ni * 16 + l15;
            const float bias = bout[n];
#pragma unroll
            for (int i = 0; i < 8; ++i) {
                int m = mbase + mi * 16 + mofs + i;
                out[(size_t)m * Vv + n] = acc[mi][ni][i] + bias;
            }
        }
    }
}

// ---------------------------------------------------------------------------
// Copy final hidden f32 [L*B*H] into the tail of d_out
// ---------------------------------------------------------------------------
__global__ void copy_f32(const float* __restrict__ src,
                         float* __restrict__ dst, int n) {
    int i = blockIdx.x * blockDim.x + threadIdx.x;
    if (i < n) dst[i] = src[i];
}

// ---------------------------------------------------------------------------
extern "C" void kernel_launch(void* const* d_in, const int* in_sizes, int n_in,
                              void* d_out, int out_size, void* d_ws, size_t ws_size,
                              hipStream_t stream) {
    const int*   tok  = (const int*)  d_in[0];   // [S,B]
    const float* h0   = (const float*)d_in[1];   // [L,B,H]
    const float* emb  = (const float*)d_in[2];   // [V,E]
    const float* Wih  = (const float*)d_in[3];   // [L,H,E]
    const float* bih  = (const float*)d_in[4];   // [L,H]
    const float* Whh  = (const float*)d_in[5];   // [L,H,H]
    const float* bhh  = (const float*)d_in[6];   // [L,H]
    const float* Wout = (const float*)d_in[7];   // [V,H]
    const float* bout = (const float*)d_in[8];   // [V]

    // Workspace carve-up (256B aligned)
    char* ws = (char*)d_ws;
    size_t off = 0;
    auto alloc = [&](size_t bytes) -> char* {
        char* p = ws + off;
        off = (off + bytes + 255) & ~(size_t)255;
        return p;
    };
    unsigned short* WihB  = (unsigned short*)alloc((size_t)Ll * Hh * Ee * 2);     //  4 MiB
    unsigned short* WhhB  = (unsigned short*)alloc((size_t)Ll * Hh * Hh * 2);     //  4 MiB
    unsigned short* WoutB = (unsigned short*)alloc((size_t)Vv * Hh * 2);          // ~20 MB
    unsigned short* X0B   = (unsigned short*)alloc((size_t)Ss * Bb * Ee * 2);     // ~17 MB
    unsigned short* H2all = (unsigned short*)alloc((size_t)Ss * Bb * Hh * 2);     // ~17 MB
    unsigned short* Hbf   = (unsigned short*)alloc((size_t)2 * Ll * Bb * Hh * 2); // ping-pong
    float*          Hf32  = (float*)         alloc((size_t)2 * Ll * Bb * Hh * 4); // ping-pong

    // Weight conversions (bf16 stays resident in 192MB L2 across step kernels)
    long nWih = (long)Ll * Hh * Ee;
    long nWhh = (long)Ll * Hh * Hh;
    long nWo  = (long)Vv * Hh;
    cvt_f32_bf16<<<(unsigned)((nWih + 255) / 256), 256, 0, stream>>>(Wih,  WihB,  nWih);
    cvt_f32_bf16<<<(unsigned)((nWhh + 255) / 256), 256, 0, stream>>>(Whh,  WhhB,  nWhh);
    cvt_f32_bf16<<<(unsigned)((nWo  + 255) / 256), 256, 0, stream>>>(Wout, WoutB, nWo);

    // Embedding lookup + scale
    long nX0 = (long)Ss * Bb * Ee;
    embed_scale<<<(unsigned)((nX0 + 255) / 256), 256, 0, stream>>>(tok, emb, X0B, nX0);

    // Hidden init (parity 0)
    int nH = Ll * Bb * Hh;
    init_hidden<<<(nH + 255) / 256, 256, 0, stream>>>(h0, Hbf, Hf32, nH);

    auto hbf_slot = [&](int p, int l) {
        return Hbf + ((size_t)(p * Ll + l)) * Bb * Hh;
    };
    auto hf_slot = [&](int p, int l) {
        return Hf32 + ((size_t)(p * Ll + l)) * Bb * Hh;
    };

    // Sequential scan over time steps; stream ordering is the dependency chain.
    for (int t = 0; t < Ss; ++t) {
        int rp = t & 1;
        int wp = rp ^ 1;
        // layer 0: x = X0[t]
        rnn_step<<<dim3(Hh / 16), 128, 0, stream>>>(
            X0B + (size_t)t * Bb * Ee, hbf_slot(rp, 0),
            WihB, WhhB, bih, bhh,
            hbf_slot(wp, 0), hf_slot(wp, 0), (unsigned short*)nullptr);
        // layer 1: x = layer-0 output of this step; also record into H2all[t]
        rnn_step<<<dim3(Hh / 16), 128, 0, stream>>>(
            hbf_slot(wp, 0), hbf_slot(rp, 1),
            WihB + (size_t)Hh * Ee, WhhB + (size_t)Hh * Hh, bih + Hh, bhh + Hh,
            hbf_slot(wp, 1), hf_slot(wp, 1), H2all + (size_t)t * Bb * Hh);
    }

    // Big output GEMM: all steps at once, register-blocked 2x5 tiles/wave.
    int wave_jobs = ((Ss * Bb / 16) / MT) * ((Vv / 16) / NT);  // 256*125 = 32000
    logits_gemm<<<wave_jobs / 8, 256, 0, stream>>>(H2all, WoutB, bout,
                                                   (float*)d_out);

    // Final hidden state (f32) -> tail of d_out. Last write parity:
    int fp = ((Ss - 1) & 1) ^ 1;                    // = 0 for S=128
    copy_f32<<<(nH + 255) / 256, 256, 0, stream>>>(
        Hf32 + (size_t)fp * Ll * Bb * Hh,
        (float*)d_out + (size_t)Ss * Bb * Vv, nH);
}
